// RNNLM_2233382994363
// MI455X (gfx1250) — matrix operand order repaired and checked by reading, not compiled, not run on hardware
//
#include <hip/hip_runtime.h>
#include <hip/hip_bf16.h>

typedef unsigned short u16;
typedef __attribute__((ext_vector_type(16))) __bf16 v16bf;
typedef __attribute__((ext_vector_type(8)))  float  v8f;

union FragBF { uint4 q[2]; v16bf v; };

__device__ __forceinline__ u16 f32_to_bf16(float f) {
  unsigned int u = __builtin_bit_cast(unsigned int, f);
  u += 0x7FFFu + ((u >> 16) & 1u);     // round-to-nearest-even
  return (u16)(u >> 16);
}
__device__ __forceinline__ float bf16_to_f32(u16 h) {
  unsigned int u = ((unsigned int)h) << 16;
  return __builtin_bit_cast(float, u);
}
__device__ __forceinline__ float sigm_(float x) { return 1.0f / (1.0f + __expf(-x)); }

// ---- fragment loaders (ISA 7.12.2 layouts) -------------------------------
// A: 16x32 bf16, tile in LDS row-major [rows][32] u16. lane m = lane&15,
// half = lane>>4. Elements j=0..7 -> K = 8*half + j ; j=8..15 -> K = 16+8*half+(j-8).
__device__ __forceinline__ v16bf load_fragA(const u16* tile, int rowBase, int lane) {
  int m = lane & 15, half = lane >> 4;
  const u16* row = tile + (rowBase + m) * 32;
  FragBF f;
  f.q[0] = *(const uint4*)(row + 8 * half);
  f.q[1] = *(const uint4*)(row + 16 + 8 * half);
  return f.v;
}
// B: 32x16 bf16 (K x N). lane n = lane&15, half = lane>>4; j=0..7 -> K = 8*half + j ; j=8..15 -> K = 16+8*half+(j-8).
// Weight tile in LDS as [N rows][32 K] u16 (weights are (N,K) row-major -> no transpose).
__device__ __forceinline__ v16bf load_fragB(const u16* tile, int rowBase, int lane) {
  int n = lane & 15, half = lane >> 4;
  const u16* row = tile + (rowBase + n) * 32;
  FragBF f;
  f.q[0] = *(const uint4*)(row + 8 * half);
  f.q[1] = *(const uint4*)(row + 16 + 8 * half);
  return f.v;
}

// ---- small utility kernels ------------------------------------------------
__global__ void __launch_bounds__(256) k_cvt(const float* __restrict__ src,
                                             u16* __restrict__ dst, long long n) {
  long long base = ((long long)blockIdx.x * 256 + threadIdx.x) * 4;
  if (base + 3 < n) {
    float4 v = *(const float4*)(src + base);
    dst[base + 0] = f32_to_bf16(v.x);
    dst[base + 1] = f32_to_bf16(v.y);
    dst[base + 2] = f32_to_bf16(v.z);
    dst[base + 3] = f32_to_bf16(v.w);
  }
}

__global__ void __launch_bounds__(256) k_bias(const float* __restrict__ a,
                                              const float* __restrict__ b,
                                              float* __restrict__ o) {
  int i = blockIdx.x * 256 + threadIdx.x;
  o[i] = a[i] + b[i];
}

__global__ void __launch_bounds__(256) k_embed(const int* __restrict__ tokens,
                                               const float* __restrict__ emb,
                                               u16* __restrict__ out) {
  int row = blockIdx.x;                       // t*B + b, 0..8191
  int tok = tokens[row];
  const float* src = emb + (size_t)tok * 1024;
  u16* dst = out + (size_t)row * 1024;
  for (int i = threadIdx.x; i < 1024; i += 256) dst[i] = f32_to_bf16(src[i]);
}

__global__ void __launch_bounds__(256) k_init(const float* __restrict__ hx_h,
                                              const float* __restrict__ hx_c,
                                              u16* h0, u16* h1, float* c0, float* c1) {
  int idx = blockIdx.x * 256 + threadIdx.x;   // 65536 = 2*32*1024
  int l = idx >> 15, r = idx & 32767;
  u16 hv = f32_to_bf16(hx_h[idx]);
  float cv = hx_c[idx];
  if (l == 0) { h0[r] = hv; c0[r] = cv; } else { h1[r] = hv; c1[r] = cv; }
}

__global__ void __launch_bounds__(256) k_final(const u16* __restrict__ h0,
                                               const u16* __restrict__ h1,
                                               const float* __restrict__ c0,
                                               const float* __restrict__ c1,
                                               float* __restrict__ dst) {
  int idx = blockIdx.x * 256 + threadIdx.x;   // 65536
  int r = idx & 32767;
  dst[idx]         = bf16_to_f32(idx < 32768 ? h0[r] : h1[r]);
  dst[65536 + idx] = (idx < 32768) ? c0[r] : c1[r];
}

// ---- fused LSTM step: gates = x@Wih^T + h@Whh^T + bias; activations; state update.
// Grid: 16 blocks, block covers h-columns [nb*64, nb*64+64) across all 4 gates.
// 8 waves: wave w -> gate g = w>>1, m-half mh = w&1 ; each wave = 16x64 (4 WMMA tiles).
__global__ void __launch_bounds__(256)
k_lstm_step(const u16* __restrict__ x, const u16* __restrict__ hprev,
            const u16* __restrict__ Wih, const u16* __restrict__ Whh,
            const float* __restrict__ bias, float* __restrict__ cstate,
            u16* __restrict__ hout, u16* __restrict__ hs1T, int t) {
  __shared__ __align__(16) u16 As[32 * 32];     // 32 rows (B) x 32 K
  __shared__ __align__(16) u16 Bs[256 * 32];    // 4 gates x 64 N rows x 32 K
  __shared__ float gbuf[32][256];               // gate values (B x 4*64)

  const int tid = threadIdx.x, lane = tid & 31, w = tid >> 5;
  const int g = w >> 1, mh = w & 1, nb = blockIdx.x;

  v8f acc[4] = {v8f{}, v8f{}, v8f{}, v8f{}};

  for (int kc = 0; kc < 64; ++kc) {             // concat K = 1024(x) + 1024(h)
    const u16* Asrc = (kc < 32) ? x : hprev;
    const u16* Wsrc = (kc < 32) ? Wih : Whh;
    const int k0 = (kc & 31) * 32;
    __syncthreads();
    if (tid < 128) {                            // A tile: 32x32
      int r = tid >> 2, ci = tid & 3;
      *(uint4*)(As + r * 32 + ci * 8) = *(const uint4*)(Asrc + r * 1024 + k0 + ci * 8);
    }
    for (int idx = tid; idx < 1024; idx += 256) {  // B tile: 256 rows x 32
      int r = idx >> 2, ci = idx & 3;
      int gg = r >> 6, nl = r & 63;
      const u16* src = Wsrc + (size_t)(gg * 1024 + nb * 64 + nl) * 1024 + k0 + ci * 8;
      *(uint4*)(Bs + r * 32 + ci * 8) = *(const uint4*)src;
    }
    __syncthreads();
    v16bf a = load_fragA(As, mh * 16, lane);
#pragma unroll
    for (int nt = 0; nt < 4; ++nt) {
      v16bf b = load_fragB(Bs, g * 64 + nt * 16, lane);
      acc[nt] = __builtin_amdgcn_wmma_f32_16x16x32_bf16(false, a, false, b,
                                                        (short)0, acc[nt], false, false);
    }
  }
  // gates -> LDS with bias
  {
    int nloc = lane & 15, half = lane >> 4;
#pragma unroll
    for (int nt = 0; nt < 4; ++nt) {
      float bv = bias[g * 1024 + nb * 64 + nt * 16 + nloc];
#pragma unroll
      for (int i = 0; i < 8; ++i) {
        int m = mh * 16 + i + 8 * half;
        gbuf[m][g * 64 + nt * 16 + nloc] = acc[nt][i] + bv;
      }
    }
  }
  __syncthreads();
  // activations + state update (PyTorch gate order i,f,g,o)
  for (int idx = tid; idx < 32 * 64; idx += 256) {
    int m = idx >> 6, nl = idx & 63;
    int colg = nb * 64 + nl;
    float iv = sigm_(gbuf[m][nl]);
    float fv = sigm_(gbuf[m][64 + nl]);
    float gv = tanhf(gbuf[m][128 + nl]);
    float ov = sigm_(gbuf[m][192 + nl]);
    float cp = cstate[m * 1024 + colg];
    float cn = fv * cp + iv * gv;
    float hv = ov * tanhf(cn);
    cstate[m * 1024 + colg] = cn;
    u16 hb = f32_to_bf16(hv);
    hout[m * 1024 + colg] = hb;
    if (hs1T) hs1T[((size_t)m * 256 + t) * 1024 + colg] = hb;  // (B,T,H) transposed store
  }
}

// ---- decoder GEMM: C(8192x32000) = A(8192x1024 bf16) @ decW^T + dec_b
// Block tile 64(M) x 128(N); 8 waves = 4M x 2N; wave = 16x64 (4 WMMA / K-chunk); K = 32 x 32.
__global__ void __launch_bounds__(256)
k_decoder(const u16* __restrict__ A, const u16* __restrict__ Bw,
          const float* __restrict__ bias, float* __restrict__ C) {
  __shared__ __align__(16) u16 As[64 * 32];
  __shared__ __align__(16) u16 Bs[128 * 32];
  const int tid = threadIdx.x, lane = tid & 31, w = tid >> 5;
  const int mw = w & 3, nw = w >> 2;
  const int m0 = blockIdx.y * 64;
  const int n0 = blockIdx.x * 128;

  v8f acc[4] = {v8f{}, v8f{}, v8f{}, v8f{}};

  for (int kc = 0; kc < 32; ++kc) {
    const int k0 = kc * 32;
    __syncthreads();
    {
      int r = tid >> 2, ci = tid & 3;
      *(uint4*)(As + r * 32 + ci * 8) = *(const uint4*)(A + (size_t)(m0 + r) * 1024 + k0 + ci * 8);
    }
#pragma unroll
    for (int j = 0; j < 2; ++j) {
      int idx = tid + j * 256, r = idx >> 2, ci = idx & 3;
      *(uint4*)(Bs + r * 32 + ci * 8) = *(const uint4*)(Bw + (size_t)(n0 + r) * 1024 + k0 + ci * 8);
    }
    if (kc + 1 < 32) {  // hint next B chunk toward L2 (global_prefetch_b8)
      int r = (tid >> 1);
      __builtin_prefetch(Bw + (size_t)(n0 + r) * 1024 + k0 + 32, 0, 0);
    }
    __syncthreads();
    v16bf a = load_fragA(As, mw * 16, lane);
#pragma unroll
    for (int nt = 0; nt < 4; ++nt) {
      v16bf b = load_fragB(Bs, nw * 64 + nt * 16, lane);
      acc[nt] = __builtin_amdgcn_wmma_f32_16x16x32_bf16(false, a, false, b,
                                                        (short)0, acc[nt], false, false);
    }
  }
  int nloc = lane & 15, half = lane >> 4;
#pragma unroll
  for (int nt = 0; nt < 4; ++nt) {
    int col = n0 + nw * 64 + nt * 16 + nloc;
    float bv = bias[col];
#pragma unroll
    for (int i = 0; i < 8; ++i) {
      int row = m0 + mw * 16 + i + 8 * half;
      C[(size_t)row * 32000 + col] = acc[nt][i] + bv;
    }
  }
}

// ---- host orchestration ---------------------------------------------------
extern "C" void kernel_launch(void* const* d_in, const int* in_sizes, int n_in,
                              void* d_out, int out_size, void* d_ws, size_t ws_size,
                              hipStream_t stream) {
  const int*   tokens = (const int*)d_in[0];
  const float* hx_h   = (const float*)d_in[1];
  const float* hx_c   = (const float*)d_in[2];
  const float* emb    = (const float*)d_in[3];
  const float* W_ih0  = (const float*)d_in[4];
  const float* W_hh0  = (const float*)d_in[5];
  const float* b_ih0  = (const float*)d_in[6];
  const float* b_hh0  = (const float*)d_in[7];
  const float* W_ih1  = (const float*)d_in[8];
  const float* W_hh1  = (const float*)d_in[9];
  const float* b_ih1  = (const float*)d_in[10];
  const float* b_hh1  = (const float*)d_in[11];
  const float* dec_W  = (const float*)d_in[12];
  const float* dec_b  = (const float*)d_in[13];
  float* out = (float*)d_out;

  char* ws = (char*)d_ws;
  size_t off = 0;
  auto alloc = [&](size_t bytes) -> char* {
    char* p = ws + off;
    off = (off + bytes + 255) & ~(size_t)255;
    return p;
  };
  u16*   embB  = (u16*)alloc((size_t)8192 * 1024 * 2);
  u16*   Wih0B = (u16*)alloc((size_t)4096 * 1024 * 2);
  u16*   Whh0B = (u16*)alloc((size_t)4096 * 1024 * 2);
  u16*   Wih1B = (u16*)alloc((size_t)4096 * 1024 * 2);
  u16*   Whh1B = (u16*)alloc((size_t)4096 * 1024 * 2);
  u16*   decWB = (u16*)alloc((size_t)32000 * 1024 * 2);
  u16*   hs1TB = (u16*)alloc((size_t)8192 * 1024 * 2);
  u16*   h0B   = (u16*)alloc((size_t)2 * 32 * 1024 * 2);  // parity double buffer
  u16*   h1B   = (u16*)alloc((size_t)2 * 32 * 1024 * 2);
  float* c0    = (float*)alloc((size_t)32 * 1024 * 4);
  float* c1    = (float*)alloc((size_t)32 * 1024 * 4);
  float* bias0 = (float*)alloc(4096 * 4);
  float* bias1 = (float*)alloc(4096 * 4);

  // one-time conversions / setup
  k_cvt<<<4096, 256, 0, stream>>>(W_ih0, Wih0B, (long long)4096 * 1024);
  k_cvt<<<4096, 256, 0, stream>>>(W_hh0, Whh0B, (long long)4096 * 1024);
  k_cvt<<<4096, 256, 0, stream>>>(W_ih1, Wih1B, (long long)4096 * 1024);
  k_cvt<<<4096, 256, 0, stream>>>(W_hh1, Whh1B, (long long)4096 * 1024);
  k_cvt<<<32000, 256, 0, stream>>>(dec_W, decWB, (long long)32000 * 1024);
  k_bias<<<16, 256, 0, stream>>>(b_ih0, b_hh0, bias0);
  k_bias<<<16, 256, 0, stream>>>(b_ih1, b_hh1, bias1);
  k_embed<<<8192, 256, 0, stream>>>(tokens, emb, embB);
  k_init<<<256, 256, 0, stream>>>(hx_h, hx_c, h0B, h1B, c0, c1);

  // sequential recurrence: 2 fused kernels per step
  const int T = 256;
  for (int t = 0; t < T; ++t) {
    const u16* x0    = embB + (size_t)t * 32 * 1024;
    const u16* h0in  = h0B + (size_t)(t & 1) * 32768;
    u16*       h0out = h0B + (size_t)((t & 1) ^ 1) * 32768;
    k_lstm_step<<<16, 256, 0, stream>>>(x0, h0in, Wih0B, Whh0B, bias0, c0, h0out, nullptr, t);
    const u16* h1in  = h1B + (size_t)(t & 1) * 32768;
    u16*       h1out = h1B + (size_t)((t & 1) ^ 1) * 32768;
    k_lstm_step<<<16, 256, 0, stream>>>(h0out, h1in, Wih1B, Whh1B, bias1, c1, h1out, hs1TB, t);
  }

  // decoder: (B*T, V) = hs1T @ dec_W^T + dec_b, written straight into d_out
  k_decoder<<<dim3(250, 128), 256, 0, stream>>>(hs1TB, decWB, dec_b, out);

  // h_n / c_n tail of d_out (T even -> final h slot is 0)
  k_final<<<256, 256, 0, stream>>>(h0B, h1B, c0, c1, out + (size_t)8192 * 32000);
}
